// PairScorer_26276609917265
// MI455X (gfx1250) — matrix operand order repaired and checked by reading, not compiled, not run on hardware
//
#include <hip/hip_runtime.h>
#include <hip/hip_bf16.h>

typedef __attribute__((ext_vector_type(16))) _Float16 v16h;
typedef __attribute__((ext_vector_type(8)))  _Float16 v8h;
typedef __attribute__((ext_vector_type(2)))  _Float16 v2h;
typedef __attribute__((ext_vector_type(8)))  float    v8f;
typedef __attribute__((ext_vector_type(4)))  unsigned v4u;
typedef __attribute__((ext_vector_type(8)))  int      v8i;
typedef __attribute__((ext_vector_type(4)))  int      v4i;

#define NSPAN 1024
#define DDIM  512
#define HDIM  150
#define HPAD  160
#define WINSZ 250
#define OUTW  (WINSZ + 1)

#define MBLK   128   // rows per workgroup (2 workgroups per span)
#define MTILES 8     // MBLK/16
#define NT     10    // HPAD/16
#define KB1    16    // DDIM/32
#define KB2    5     // HPAD/32

// workspace layout (bytes)
#define OFF_B1 0                          // KB1*NT*32*16 halves = 163840 B
#define OFF_B2 (OFF_B1 + 163840)          // KB2*NT*32*16 halves = 51200 B
#define OFF_W3 (OFF_B2 + 51200)           // 160 f32 = 640 B
#define OFF_GI (OFF_W3 + 1024)            // NSPAN*HPAD f32 = 655360 B
#define OFF_GJ (OFF_GI + 655360)          // NSPAN*HPAD f32 = 655360 B

// LDS carve (bytes); one k-chunk of B1 fragments = NT*512 halves = 10240 B
#define XHALVES   4096                    // MTILES*32*16 per X buffer
#define B1HALVES  5120                    // NT*32*16 per B1 buffer
#define LDSX_OFF  0                       // 2 * 8192 B
#define LDSB1_OFF 16384                   // 2 * 10240 B
#define LDSH1_OFF 36864                   // 40960 B
#define LDS_TOTAL 77824

// A-fragment K position for element e=2v+t, lane-half h (16-bit A, 16x32):
//   v<4 : K = 2v + 8h + t ; v>=4 : K = 16 + 2(v-4) + 8h + t
// B-fragment K position (16-bit B, 32x16): same as A (v<4: K = 2v + 8h + t ; v>=4: K = 16 + 2(v-4) + 8h + t)

__global__ __launch_bounds__(256)
void prep_weights_kernel(const float* __restrict__ W1, const float* __restrict__ W2,
                         const float* __restrict__ W3,
                         _Float16* __restrict__ B1, _Float16* __restrict__ B2,
                         float* __restrict__ w3p)
{
    int idx = blockIdx.x * 256 + threadIdx.x;
    const int T1 = KB1 * NT * 32 * 16;   // 81920
    const int T2 = KB2 * NT * 32 * 16;   // 25600
    if (idx < T1) {
        int e = idx & 15, L = (idx >> 4) & 31, bidx = idx >> 9;
        int nt = bidx % NT, kb = bidx / NT;
        int hf = L >> 4;
        int n  = (nt << 4) + (L & 15);
        int v  = e >> 1, t = e & 1;
        int k  = (kb << 5) + ((e >> 3) << 4) + (hf << 3) + (e & 7);     // B-fragment pattern
        float val = (n < HDIM) ? W1[(2 * DDIM + k) * HDIM + n] : 0.f;  // W1c rows
        B1[idx] = (_Float16)val;
    } else if (idx < T1 + T2) {
        int id = idx - T1;
        int e = id & 15, L = (id >> 4) & 31, bidx = id >> 9;
        int nt = bidx % NT, kb = bidx / NT;
        int hf = L >> 4;
        int n  = (nt << 4) + (L & 15);
        int v  = e >> 1, t = e & 1;
        int k  = (kb << 5) + ((e >> 3) << 4) + (hf << 3) + (e & 7);
        float val = (k < HDIM && n < HDIM) ? W2[k * HDIM + n] : 0.f;
        B2[id] = (_Float16)val;
    } else if (idx < T1 + T2 + HPAD) {
        int n = idx - T1 - T2;
        w3p[n] = (n < HDIM) ? W3[n] : 0.f;
    }
}

__global__ __launch_bounds__(256)
void prep_terms_kernel(const float* __restrict__ g, const float* __restrict__ W1,
                       float* __restrict__ giT, float* __restrict__ gjT)
{
    int idx = blockIdx.x * 256 + threadIdx.x;
    if (idx >= NSPAN * HPAD) return;
    int r = idx / HPAD, n = idx % HPAD;
    float sa = 0.f, sb = 0.f;
    if (n < HDIM) {
        const float* gr = g + r * DDIM;
        for (int k = 0; k < DDIM; ++k) {
            float gv = gr[k];
            sa = fmaf(gv, W1[k * HDIM + n], sa);             // W1a
            sb = fmaf(gv, W1[(DDIM + k) * HDIM + n], sb);    // W1b
        }
    }
    giT[idx] = sa;
    gjT[idx] = sb;
}

// Tensor Data Mover: 1-D contiguous copy of nbytes8*8 bytes global -> LDS.
// D# packing per CDNA5 ISA (group0: count/lds/global/type,
// group1: data_size=8B, tensor_dim0=tile_dim0=nbytes8, stride0=nbytes8).
// This toolchain exposes the 6-arg builtin; trailing descriptor groups are
// unused for a <=2-D tensor and passed zero-filled.
__device__ __forceinline__ void tdm_copy_1d(unsigned long long gaddr, unsigned lds_addr,
                                            unsigned nbytes8)
{
    v4u g0 = { 1u,                                            // count=1
               lds_addr,                                      // D#.lds_addr
               (unsigned)(gaddr & 0xffffffffull),             // global_addr[31:0]
               (unsigned)((gaddr >> 32) & 0x01ffffffull) | (2u << 30) }; // ga[56:32] | type=2
    v8i g1 = { (int)0x30000u,                                 // data_size=3 (8B units)
               (int)(nbytes8 << 16),                          // tensor_dim0[15:0]
               (int)(1u << 16),                               // tensor_dim0 hi | tensor_dim1=1
               (int)(nbytes8 << 16),                          // tile_dim0
               0,                                             // tile_dim1=0 (unused), tile_dim2=0
               (int)nbytes8,                                  // tensor_dim0_stride[31:0]
               0, 0 };
    v4i gz4 = { 0, 0, 0, 0 };
    v8i gz8 = { 0, 0, 0, 0, 0, 0, 0, 0 };
    __builtin_amdgcn_tensor_load_to_lds(g0, g1, gz4, gz4, gz8, 0);
}

// prefetch next chunk's g_j data (kept off the WMMA wait path)
#define LOAD_GJ(kb_) do {                                                      \
    const int kc_ = (kb_) << 5;                                                \
    _Pragma("unroll")                                                          \
    for (int it = 0; it < 8; ++it)                                             \
        pgj[it] = *reinterpret_cast<const float2*>(&g[goffj[it] + kc_]);       \
} while (0)

// convert + scatter the prefetched chunk into A-fragment layout in LDS
#define STORE_X(buf_, kb_) do {                                                \
    const int kc_ = (kb_) << 5;                                                \
    _Pragma("unroll")                                                          \
    for (int it = 0; it < 8; ++it) {                                           \
        int f_  = it * 256 + tid;                                              \
        int ml_ = f_ >> 4;                                                     \
        int kl_ = (f_ & 15) << 1;                                              \
        int m_  = mbase + ml_;                                                 \
        int j_  = i - WINSZ + m_;                                              \
        float msk_ = (j_ >= 0 && m_ < WINSZ) ? 1.f : 0.f;                      \
        float2 gi2_ = *reinterpret_cast<const float2*>(&g[i * DDIM + kl_ + kc_]); \
        float x0_ = gi2_.x * pgj[it].x * msk_;                                 \
        float x1_ = gi2_.y * pgj[it].y * msk_;                                 \
        int hi_ = kl_ >> 4, rem_ = kl_ & 15, hf_ = rem_ >> 3, r_ = rem_ & 7;   \
        int v_  = (hi_ << 2) + (r_ >> 1);                                      \
        int mt_ = ml_ >> 4, mr_ = ml_ & 15;                                    \
        int L_  = (hf_ << 4) + mr_;                                            \
        int sidx_ = ((mt_ << 5) + L_) * 16 + (v_ << 1);                        \
        v2h pk_ = { (_Float16)x0_, (_Float16)x1_ };                            \
        *reinterpret_cast<v2h*>(&ldsX[(buf_) * XHALVES + sidx_]) = pk_;        \
    }                                                                          \
} while (0)

__global__ __launch_bounds__(256)
void pair_main_kernel(const float* __restrict__ g, const float* __restrict__ sm,
                      const float* __restrict__ b1, const float* __restrict__ b2,
                      const float* __restrict__ b3,
                      const _Float16* __restrict__ B1, const _Float16* __restrict__ B2,
                      const float* __restrict__ w3p,
                      const float* __restrict__ giT, const float* __restrict__ gjT,
                      float* __restrict__ out)
{
    __shared__ __align__(128) char smem[LDS_TOTAL];
    _Float16* ldsX  = (_Float16*)(smem + LDSX_OFF);    // 2 x 4096 halves: X double buffer
    _Float16* ldsB1 = (_Float16*)(smem + LDSB1_OFF);   // 2 x 5120 halves: B1 chunk (TDM dest)
    _Float16* ldsH1 = (_Float16*)(smem + LDSH1_OFF);   // 20480 halves: hidden1 fragments

    const int blk   = blockIdx.x;
    const int i     = blk >> 1;              // span index
    const int mbase = (blk & 1) * MBLK;      // 0 or 128 within the 256-row padded window
    const int tid   = threadIdx.x;
    const int wave  = tid >> 5;              // local M-tile 0..7
    const int lane  = tid & 31;
    const int half  = lane >> 4;
    const int lrow  = lane & 15;

    // LDS byte address of the B1 staging buffers (flat-address low 32 bits == LDS offset)
    const unsigned ldsB1_addr = (unsigned)(size_t)(smem + LDSB1_OFF);
    const unsigned long long b1_gbase = (unsigned long long)(size_t)B1;

    const v8f zero = {0.f, 0.f, 0.f, 0.f, 0.f, 0.f, 0.f, 0.f};
    v8f acc[NT];
#pragma unroll
    for (int nt = 0; nt < NT; ++nt) acc[nt] = zero;

    // per-thread invariant g_j offsets for the cooperative fill (clamped in-bounds)
    int goffj[8];
#pragma unroll
    for (int it = 0; it < 8; ++it) {
        int f  = it * 256 + tid;
        int ml = f >> 4;
        int kl = (f & 15) << 1;
        int m  = mbase + ml;
        int j  = i - WINSZ + m;
        int jc = (j >= 0 && m < WINSZ) ? j : 0;
        goffj[it] = jc * DDIM + kl;
    }

    float2 pgj[8];

    // ---------------- GEMM1: X[m][k] = g_i[k]*g_j[k]  (K = 512) ----------------
    if (wave == 0) tdm_copy_1d(b1_gbase, ldsB1_addr, 1280u);   // chunk 0 -> buffer 0
    LOAD_GJ(0);
    STORE_X(0, 0);
    if (wave == 0) __builtin_amdgcn_s_wait_tensorcnt(0);
    __syncthreads();

    for (int kb = 0; kb < KB1; ++kb) {
        const int cur = kb & 1;
        // kick off the TDM copy of the next B1 chunk into the other buffer
        if (wave == 0 && kb + 1 < KB1)
            tdm_copy_1d(b1_gbase + (unsigned long long)(kb + 1) * (NT * 512 * 2),
                        ldsB1_addr + (unsigned)(cur ^ 1) * (B1HALVES * 2), 1280u);
        // A fragment for this wave's M-tile
        const v8h* pa = reinterpret_cast<const v8h*>(&ldsX[cur * XHALVES + ((wave << 5) + lane) * 16]);
        v8h a0 = pa[0], a1 = pa[1];
        v16h afrag = __builtin_shufflevector(a0, a1, 0,1,2,3,4,5,6,7,8,9,10,11,12,13,14,15);
        if (kb + 1 < KB1) LOAD_GJ(kb + 1);   // global prefetch overlaps the WMMAs below
#pragma unroll
        for (int nt = 0; nt < NT; ++nt) {
            const v8h* pb = reinterpret_cast<const v8h*>(
                &ldsB1[cur * B1HALVES + ((nt << 5) + lane) * 16]);
            v8h bl = pb[0], bh = pb[1];
            v16h bfrag = __builtin_shufflevector(bl, bh, 0,1,2,3,4,5,6,7,8,9,10,11,12,13,14,15);
            acc[nt] = __builtin_amdgcn_wmma_f32_16x16x32_f16(
                false, afrag, false, bfrag, (short)0, acc[nt], false, false);
        }
        if (kb + 1 < KB1) STORE_X(cur ^ 1, kb + 1);
        if (wave == 0 && kb + 1 < KB1) __builtin_amdgcn_s_wait_tensorcnt(0);
        __syncthreads();
    }

    // ------------- epilogue 1: h1 = relu(acc + giT[i] + gjT[j] + b1) -> ldsH1 -------------
#pragma unroll
    for (int nt = 0; nt < NT; ++nt) {
        int n = (nt << 4) + lrow;
        float giv = giT[i * HPAD + n];
        float b1v = (n < HDIM) ? b1[n] : 0.f;
        // position of column n inside GEMM2 A fragments (A pattern)
        int kb2 = n >> 5, kl = n & 31;
        int hi = kl >> 4, rem = kl & 15, hf = rem >> 3, r = rem & 7;
        int v2i = (hi << 2) + (r >> 1), t2 = r & 1;
        int e2  = (v2i << 1) + t2;
#pragma unroll
        for (int v = 0; v < 8; ++v) {
            int mloc = (wave << 4) + v + (half << 3);
            int m = mbase + mloc;
            int j = i - WINSZ + m;
            int jc = j < 0 ? 0 : (j > NSPAN - 1 ? NSPAN - 1 : j);
            float hv = acc[nt][v] + giv + gjT[jc * HPAD + n] + b1v;
            hv = fmaxf(hv, 0.f);
            if (n >= HDIM) hv = 0.f;   // padded columns feed GEMM2 K-pad: force 0
            int L2 = (hf << 4) + (mloc & 15);
            ldsH1[(((wave * KB2 + kb2) << 5) + L2) * 16 + e2] = (_Float16)hv;
        }
    }
    __syncthreads();

    // ---------------- GEMM2: h2 = relu(h1 @ W2 + b2)  (K = 160 padded) ----------------
    v8f acc2[NT];
#pragma unroll
    for (int nt = 0; nt < NT; ++nt) acc2[nt] = zero;
#pragma unroll
    for (int kb = 0; kb < KB2; ++kb) {
        const v8h* pa = reinterpret_cast<const v8h*>(
            &ldsH1[(((wave * KB2 + kb) << 5) + lane) * 16]);
        v8h a0 = pa[0], a1 = pa[1];
        v16h afrag = __builtin_shufflevector(a0, a1, 0,1,2,3,4,5,6,7,8,9,10,11,12,13,14,15);
#pragma unroll
        for (int nt = 0; nt < NT; ++nt) {
            const v8h* pb = reinterpret_cast<const v8h*>(&B2[(((kb * NT + nt) << 5) + lane) * 16]);
            v8h bl = pb[0], bh = pb[1];
            v16h bfrag = __builtin_shufflevector(bl, bh, 0,1,2,3,4,5,6,7,8,9,10,11,12,13,14,15);
            acc2[nt] = __builtin_amdgcn_wmma_f32_16x16x32_f16(
                false, afrag, false, bfrag, (short)0, acc2[nt], false, false);
        }
    }

    // ------------- epilogue 2: s = h2 . W3 ; out = sm_i + sm_j + s + b3 -------------
    float part[8] = {0.f, 0.f, 0.f, 0.f, 0.f, 0.f, 0.f, 0.f};
#pragma unroll
    for (int nt = 0; nt < NT; ++nt) {
        int n = (nt << 4) + lrow;
        float b2v = (n < HDIM) ? b2[n] : 0.f;
        float w3v = w3p[n];                       // 0 for padded columns
#pragma unroll
        for (int v = 0; v < 8; ++v) {
            float h2 = fmaxf(acc2[nt][v] + b2v, 0.f);
            part[v] += h2 * w3v;
        }
    }
    // reduce over the 16 column-lanes of each wave-half (wave32: masks<16 stay in-half)
#pragma unroll
    for (int v = 0; v < 8; ++v) {
        part[v] += __shfl_xor(part[v], 1, 32);
        part[v] += __shfl_xor(part[v], 2, 32);
        part[v] += __shfl_xor(part[v], 4, 32);
        part[v] += __shfl_xor(part[v], 8, 32);
    }
    float smi = sm[i];
    float b3v = b3[0];
#pragma unroll
    for (int v = 0; v < 8; ++v) {
        if (lrow == v) {
            int m = mbase + (wave << 4) + (half << 3) + v;
            if (m < WINSZ) {
                int j = i - WINSZ + m;
                float val = 0.f;
                if (j >= 0) val = smi + sm[j] + part[v] + b3v;
                out[i * OUTW + m] = val;
            }
        }
    }
    if (mbase == 0 && tid == 0) out[i * OUTW + WINSZ] = 0.f;
}

extern "C" void kernel_launch(void* const* d_in, const int* in_sizes, int n_in,
                              void* d_out, int out_size, void* d_ws, size_t ws_size,
                              hipStream_t stream) {
    (void)in_sizes; (void)n_in; (void)out_size; (void)ws_size;
    const float* g  = (const float*)d_in[0];
    const float* sm = (const float*)d_in[1];
    const float* W1 = (const float*)d_in[2];
    const float* b1 = (const float*)d_in[3];
    const float* W2 = (const float*)d_in[4];
    const float* b2 = (const float*)d_in[5];
    const float* W3 = (const float*)d_in[6];
    const float* b3 = (const float*)d_in[7];
    char* ws = (char*)d_ws;
    _Float16* B1 = (_Float16*)(ws + OFF_B1);
    _Float16* B2 = (_Float16*)(ws + OFF_B2);
    float* w3p = (float*)(ws + OFF_W3);
    float* giT = (float*)(ws + OFF_GI);
    float* gjT = (float*)(ws + OFF_GJ);
    float* out = (float*)d_out;

    int prepw_elems = KB1 * NT * 32 * 16 + KB2 * NT * 32 * 16 + HPAD;
    prep_weights_kernel<<<(prepw_elems + 255) / 256, 256, 0, stream>>>(W1, W2, W3, B1, B2, w3p);
    prep_terms_kernel<<<(NSPAN * HPAD + 255) / 256, 256, 0, stream>>>(g, W1, giT, gjT);
    pair_main_kernel<<<NSPAN * 2, 256, 0, stream>>>(g, sm, b1, b2, b3, B1, B2, w3p, giT, gjT, out);
}
